// CNNMambaBranch_87179246174646
// MI455X (gfx1250) — matrix operand order repaired and checked by reading, not compiled, not run on hardware
//
#include <hip/hip_runtime.h>
#include <stdint.h>

// ---------------- types ----------------
typedef __attribute__((ext_vector_type(16))) __bf16 v16bf;
typedef __attribute__((ext_vector_type(8)))  float  v8f;
typedef unsigned int u32x4 __attribute__((ext_vector_type(4)));
typedef int          i32x4 __attribute__((ext_vector_type(4)));
typedef int          i32x8 __attribute__((ext_vector_type(8)));

#define B_SZ     16
#define L_SEQ    4096
#define DMODEL   128
#define DINNER   256
#define DSTATE   16
#define DTRANK   8
#define NTOK     (B_SZ * L_SEQ)          // 65536
#define XDBL_N   (DTRANK + 2 * DSTATE)   // 40
#define APAD     8                       // LDS row padding (elements = 16B = 4 DWORDs)

// ---------------- scalar helpers ----------------
__device__ __forceinline__ float bf2f(unsigned short h) {
    return __uint_as_float(((unsigned int)h) << 16);
}
__device__ __forceinline__ unsigned short f2bf(float f) {
    unsigned int u = __float_as_uint(f);
    u += 0x7FFFu + ((u >> 16) & 1u);   // round-to-nearest-even
    return (unsigned short)(u >> 16);
}

// ---------------- WMMA fragment loaders ----------------
// A fragment: 16x32 bf16, row-major source, stride lda (elements).
// lanes 0-15: rows M=0..15, K = 0..15; lanes 16-31: same rows, K = 16..31.
__device__ __forceinline__ v16bf load_a_frag(const unsigned short* A, int lda) {
    int lane  = threadIdx.x & 31;
    int row   = lane & 15;
    int khalf = lane >> 4;
    union { v16bf v; unsigned int u[8]; } r;
    const unsigned short* base = A + (size_t)row * lda;
#pragma unroll
    for (int v = 0; v < 8; ++v) {
        int k = khalf * 16 + v * 2;
        r.u[v] = *reinterpret_cast<const unsigned int*>(base + k);
    }
    return r.v;
}

// B fragment: 32x16 bf16 from row-major N x K weight W[n*ldk + k], B(k,n)=W[n,k].
__device__ __forceinline__ v16bf load_b_frag(const unsigned short* W, int ldk) {
    int lane  = threadIdx.x & 31;
    int n     = lane & 15;
    int kbase = (lane >> 4) * 16;
    union { v16bf v; unsigned int u[8]; } r;
    const unsigned short* base = W + (size_t)n * ldk + kbase;
#pragma unroll
    for (int v = 0; v < 8; ++v)
        r.u[v] = *reinterpret_cast<const unsigned int*>(base + 2 * v);
    return r.v;
}

// Guarded B fragment: rows n0+n beyond nmax read as zero (for N=40 padded to 48).
__device__ __forceinline__ v16bf load_b_frag_guard(const unsigned short* W, int ldk,
                                                   int n0, int nmax) {
    int lane  = threadIdx.x & 31;
    int n     = (lane & 15) + n0;
    int kbase = (lane >> 4) * 16;
    union { v16bf v; unsigned int u[8]; } r;
    const unsigned short* base = W + (size_t)n * ldk + kbase;
    bool ok = (n < nmax);
#pragma unroll
    for (int v = 0; v < 8; ++v)
        r.u[v] = ok ? *reinterpret_cast<const unsigned int*>(base + 2 * v) : 0u;
    return r.v;
}

__device__ __forceinline__ v8f wmma_bf16(v16bf a, v16bf b, v8f c) {
    return __builtin_amdgcn_wmma_f32_16x16x32_bf16(false, a, false, b, (short)0, c,
                                                   false, false);
}

// ---------------- TDM (Tensor Data Mover) staging: global -> LDS ----------------
#if __has_builtin(__builtin_amdgcn_tensor_load_to_lds) && __has_builtin(__builtin_amdgcn_s_wait_tensorcnt)
#define HAVE_TDM 1
#else
#define HAVE_TDM 0
#endif

#if HAVE_TDM
// 2D tile: rows x cols bf16 elements, source row stride == cols, LDS gets
// pad_amount=4 DWORDs (=APAD elements) after every row (pad_interval code).
__device__ __forceinline__ void tdm_stage_2d(const unsigned short* gsrc,
                                             unsigned int lds_off,
                                             int cols, int rows, int pad_code) {
    unsigned long long ga = (unsigned long long)(uintptr_t)gsrc;
    u32x4 g0;
    g0[0] = 1u;                                               // count=1, user mode
    g0[1] = lds_off;                                          // lds_addr (bytes)
    g0[2] = (unsigned int)ga;                                 // global_addr[31:0]
    g0[3] = (unsigned int)((ga >> 32) & 0x01FFFFFFull)        // global_addr[56:32]
          | (2u << 30);                                       // type=2 ("image")
    unsigned int td0 = (unsigned int)cols;                    // tensor_dim0 == stride
    i32x8 g1;
    g1[0] = (int)((1u << 16)                                  // data_size = 2 bytes
                | (1u << 20)                                  // pad_enable
                | ((unsigned int)pad_code << 22)              // pad_interval code
                | (3u << 25));                                // pad_amount = 4 DWORDs
    g1[1] = (int)((td0 & 0xFFFFu) << 16);                     // tensor_dim0 lo16
    g1[2] = (int)(((td0 >> 16) & 0xFFFFu)
                | (((unsigned int)rows & 0xFFFFu) << 16));    // tensor_dim1 lo16
    g1[3] = (int)(((unsigned int)cols & 0xFFFFu) << 16);      // tile_dim0
    g1[4] = (int)((unsigned int)rows & 0xFFFFu);              // tile_dim1 (tile_dim2=0)
    g1[5] = (int)td0;                                         // tensor_dim0_stride lo32
    g1[6] = 0;
    g1[7] = 0;
    i32x4 gz = {0, 0, 0, 0};
#if defined(__clang_major__) && __clang_major__ >= 23
    i32x8 gz8 = {0, 0, 0, 0, 0, 0, 0, 0};
    __builtin_amdgcn_tensor_load_to_lds(g0, g1, gz, gz, gz8, 0);
#else
    __builtin_amdgcn_tensor_load_to_lds(g0, g1, gz, gz, 0);
#endif
}
#endif

// Stage rows x cols bf16 A-tile into padded LDS (row stride cols+APAD), then barrier.
__device__ __forceinline__ void stage_a_tile(const unsigned short* gsrc,
                                             unsigned short* smem,
                                             int cols, int rows, int pad_code) {
#if HAVE_TDM
    if (threadIdx.x < 32) {                       // one wave issues the TDM op
        tdm_stage_2d(gsrc, (unsigned int)(uintptr_t)smem, cols, rows, pad_code);
        __builtin_amdgcn_s_wait_tensorcnt(0);
    }
#else
    int chunks = rows * cols / 8;                 // 16B chunks
    int cpr    = cols / 8;
    for (int c = threadIdx.x; c < chunks; c += blockDim.x) {
        int r = c / cpr, cb = c - r * cpr;
        *reinterpret_cast<uint4*>(smem + (size_t)r * (cols + APAD) + cb * 8) =
            *reinterpret_cast<const uint4*>(gsrc + (size_t)r * cols + cb * 8);
    }
#endif
    __syncthreads();
}

// ---------------- K0: weight f32 -> bf16 ----------------
__global__ __launch_bounds__(256) void k_wconv(const float* w_in, const float* w_xp,
                                               const float* w_out, unsigned short* o_in,
                                               unsigned short* o_xp, unsigned short* o_out) {
    int i = blockIdx.x * 256 + threadIdx.x;
    if (i < 512 * 128)  o_in[i]  = f2bf(w_in[i]);
    if (i < 40 * 256)   o_xp[i]  = f2bf(w_xp[i]);
    if (i < 128 * 256)  o_out[i] = f2bf(w_out[i]);
}

// ---------------- K1: conv1d(k=3,pad=1) + BN + ReLU -> token-major bf16 ----------------
__global__ __launch_bounds__(256) void k_front(const float* __restrict__ x,
                                               const float* __restrict__ cw,
                                               const float* __restrict__ cb,
                                               const float* __restrict__ g,
                                               const float* __restrict__ be,
                                               const float* __restrict__ mean,
                                               const float* __restrict__ var,
                                               unsigned short* __restrict__ hA) {
    size_t idx = (size_t)blockIdx.x * 256 + threadIdx.x;
    if (idx >= (size_t)NTOK * DMODEL) return;
    int    c   = (int)(idx & (DMODEL - 1));
    size_t tok = idx >> 7;
    int    l   = (int)(tok & (L_SEQ - 1));
    int    b   = (int)(tok >> 12);
    const float* xb = x + (size_t)b * L_SEQ;
    float acc = cb[c];
    if (l > 0)          acc += xb[l - 1] * cw[c * 3 + 0];
    acc += xb[l] * cw[c * 3 + 1];
    if (l < L_SEQ - 1)  acc += xb[l + 1] * cw[c * 3 + 2];
    acc = (acc - mean[c]) * rsqrtf(var[c] + 1e-5f);
    acc = acc * g[c] + be[c];
    acc = fmaxf(acc, 0.0f);
    hA[idx] = f2bf(acc);
}

// ---------------- K2: in_proj GEMM (M=65536,K=128,N=512) -> U | Z ----------------
// grid (2048, 2): 32-row M macro-tile staged once via TDM; blockIdx.y 0->U, 1->Z.
// Each of 8 waves computes a 2x2 grid of 16x16 tiles: 4 WMMAs per load group.
__global__ __launch_bounds__(256) void k_inproj(const unsigned short* __restrict__ A,
                                                const unsigned short* __restrict__ W,
                                                unsigned short* __restrict__ U,
                                                unsigned short* __restrict__ Z) {
    __shared__ unsigned short sA[32 * (DMODEL + APAD)];
    int mtile = blockIdx.x;                       // 0..2047 (32-row tiles)
    int wave  = threadIdx.x >> 5;

    stage_a_tile(A + (size_t)mtile * 32 * DMODEL, sA, DMODEL, 32, 5 /*64 DW rows*/);

    int ntile = (int)blockIdx.y * 16 + wave * 2;
    const unsigned short* W0 = W + (size_t)ntile * 16 * DMODEL;
    const unsigned short* W1 = W0 + (size_t)16 * DMODEL;
    const int ldsa = DMODEL + APAD;
    v8f c00 = {}, c01 = {}, c10 = {}, c11 = {};
#pragma unroll
    for (int kk = 0; kk < DMODEL; kk += 32) {
        v16bf a0 = load_a_frag(sA + kk, ldsa);                 // rows 0..15 (LDS)
        v16bf a1 = load_a_frag(sA + 16 * ldsa + kk, ldsa);     // rows 16..31 (LDS)
        v16bf b0 = load_b_frag(W0 + kk, DMODEL);
        v16bf b1 = load_b_frag(W1 + kk, DMODEL);
        c00 = wmma_bf16(a0, b0, c00);
        c01 = wmma_bf16(a0, b1, c01);
        c10 = wmma_bf16(a1, b0, c10);
        c11 = wmma_bf16(a1, b1, c11);
    }
    int lane = threadIdx.x & 31;
    int nl   = lane & 15;
    int kh   = lane >> 4;
    unsigned short* dst = (blockIdx.y == 0) ? U : Z;          // block-uniform
    int ng0 = wave * 32 + nl;                                 // 0..255 within dst
#pragma unroll
    for (int v = 0; v < 8; ++v) {
        int m0 = mtile * 32 + v + 8 * kh;                     // rows 0..15 of tile
        int m1 = m0 + 16;                                     // rows 16..31
        dst[(size_t)m0 * DINNER + ng0]      = f2bf(c00[v]);
        dst[(size_t)m0 * DINNER + ng0 + 16] = f2bf(c01[v]);
        dst[(size_t)m1 * DINNER + ng0]      = f2bf(c10[v]);
        dst[(size_t)m1 * DINNER + ng0 + 16] = f2bf(c11[v]);
    }
}

// ---------------- K3: causal depthwise conv (k=4) + SiLU ----------------
__global__ __launch_bounds__(256) void k_dwconv(const unsigned short* __restrict__ U,
                                                const float* __restrict__ w,
                                                const float* __restrict__ bws,
                                                unsigned short* __restrict__ UC) {
    size_t idx = (size_t)blockIdx.x * 256 + threadIdx.x;
    if (idx >= (size_t)NTOK * DINNER) return;
    int    d   = (int)(idx & (DINNER - 1));
    size_t tok = idx >> 8;
    int    l   = (int)(tok & (L_SEQ - 1));
    float acc = bws[d];
#pragma unroll
    for (int k = 0; k < 4; ++k) {
        int ll = l - 3 + k;
        if (ll >= 0)
            acc += bf2f(U[(tok - (size_t)(3 - k)) * DINNER + d]) * w[d * 4 + k];
    }
    float s = 1.0f / (1.0f + __expf(-acc));
    UC[idx] = f2bf(acc * s);
}

// ---------------- K4: x_proj GEMM (N=40 pad 48) -> x_dbl f32 ----------------
__global__ __launch_bounds__(96) void k_xproj(const unsigned short* __restrict__ UC,
                                              const unsigned short* __restrict__ W,
                                              float* __restrict__ XD) {
    __shared__ unsigned short sA[16 * (DINNER + APAD)];
    int mtile = blockIdx.x;
    int wave  = threadIdx.x >> 5;           // 0..2
    int n0    = wave * 16;

    stage_a_tile(UC + (size_t)mtile * 16 * DINNER, sA, DINNER, 16, 6 /*128 DW rows*/);

    v8f c = {};
#pragma unroll
    for (int kk = 0; kk < DINNER; kk += 32) {
        v16bf a  = load_a_frag(sA + kk, DINNER + APAD);
        v16bf bf = load_b_frag_guard(W + kk, DINNER, n0, XDBL_N);
        c = wmma_bf16(a, bf, c);
    }
    int lane = threadIdx.x & 31;
    int nl   = lane & 15;
    int kh   = lane >> 4;
    int ng   = n0 + nl;
    if (ng < XDBL_N) {
#pragma unroll
        for (int v = 0; v < 8; ++v) {
            int mg = mtile * 16 + v + 8 * kh;
            XD[(size_t)mg * XDBL_N + ng] = c[v];
        }
    }
}

// ---------------- K5: dt_proj + softplus + selective scan + SiLU gate ----------------
__global__ __launch_bounds__(256) void k_scan(const float* __restrict__ XD,
                                              const unsigned short* __restrict__ UC,
                                              const unsigned short* __restrict__ Zb,
                                              const float* __restrict__ dtw,
                                              const float* __restrict__ dtb,
                                              const float* __restrict__ A_log,
                                              const float* __restrict__ Dp,
                                              unsigned short* __restrict__ Y) {
    int b = blockIdx.x;             // 0..15
    int d = threadIdx.x;            // 0..255
    __shared__ float sx[2][XDBL_N];

    float h[DSTATE];
    float Ar[DSTATE];
#pragma unroll
    for (int s = 0; s < DSTATE; ++s) { h[s] = 0.0f; Ar[s] = -__expf(A_log[d * DSTATE + s]); }
    float wdt[DTRANK];
#pragma unroll
    for (int r = 0; r < DTRANK; ++r) wdt[r] = dtw[d * DTRANK + r];
    float bias = dtb[d];
    float Dd   = Dp[d];

    const float*          xd = XD + (size_t)b * L_SEQ * XDBL_N;
    const unsigned short* ud = UC + (size_t)b * L_SEQ * DINNER + d;
    const unsigned short* zd = Zb + (size_t)b * L_SEQ * DINNER + d;
    unsigned short*       yd = Y  + (size_t)b * L_SEQ * DINNER + d;

    for (int l = 0; l < L_SEQ; ++l) {
        int buf = l & 1;
        if (threadIdx.x < XDBL_N) sx[buf][threadIdx.x] = xd[(size_t)l * XDBL_N + threadIdx.x];
        if (l + 16 < L_SEQ) {   // global_prefetch_b8 a few steps ahead
            __builtin_prefetch(xd + (size_t)(l + 16) * XDBL_N, 0, 1);
            __builtin_prefetch(ud + (size_t)(l + 16) * DINNER, 0, 1);
            __builtin_prefetch(zd + (size_t)(l + 16) * DINNER, 0, 1);
        }
        __syncthreads();

        float dtr = bias;
#pragma unroll
        for (int r = 0; r < DTRANK; ++r) dtr += sx[buf][r] * wdt[r];
        float dt = (dtr > 15.0f) ? dtr : __logf(1.0f + __expf(dtr));   // softplus

        float uu  = bf2f(ud[(size_t)l * DINNER]);
        float zz  = bf2f(zd[(size_t)l * DINNER]);
        float dtu = dt * uu;
        float y   = 0.0f;
#pragma unroll
        for (int s = 0; s < DSTATE; ++s) {
            h[s] = __expf(dt * Ar[s]) * h[s] + dtu * sx[buf][DTRANK + s];
            y   += h[s] * sx[buf][DTRANK + DSTATE + s];
        }
        y += uu * Dd;
        float sig = 1.0f / (1.0f + __expf(-zz));
        y *= zz * sig;
        yd[(size_t)l * DINNER] = f2bf(y);
    }
}

// ---------------- K6: zero output accumulator ----------------
__global__ __launch_bounds__(256) void k_zero(float* out) {
    int i = blockIdx.x * 256 + threadIdx.x;
    if (i < B_SZ * DMODEL) out[i] = 0.0f;
}

// ---------------- K7: out_proj GEMM + LayerNorm + mean-over-L (atomic) ----------------
__global__ __launch_bounds__(256) void k_outln(const unsigned short* __restrict__ Y,
                                               const unsigned short* __restrict__ W,
                                               const float* __restrict__ lng,
                                               const float* __restrict__ lnb,
                                               float* __restrict__ out) {
    int mtile = blockIdx.x;                 // 0..4095 (256 tiles per batch)
    int wave  = threadIdx.x >> 5;           // 0..7 -> N tile
    __shared__ unsigned short sA[16 * (DINNER + APAD)];
    __shared__ float tile[16][132];
    __shared__ float mu[16], rs[16];
    __shared__ float part[256];

    stage_a_tile(Y + (size_t)mtile * 16 * DINNER, sA, DINNER, 16, 6);

    const unsigned short* Wb = W + (size_t)wave * 16 * DINNER;
    v8f c = {};
#pragma unroll
    for (int kk = 0; kk < DINNER; kk += 32) {
        v16bf a  = load_a_frag(sA + kk, DINNER + APAD);
        v16bf bf = load_b_frag(Wb + kk, DINNER);
        c = wmma_bf16(a, bf, c);
    }
    int lane = threadIdx.x & 31;
    int nl   = lane & 15;
    int kh   = lane >> 4;
    int ng   = wave * 16 + nl;
#pragma unroll
    for (int v = 0; v < 8; ++v) tile[v + 8 * kh][ng] = c[v];
    __syncthreads();

    if (threadIdx.x < 16) {
        int r = threadIdx.x;
        float s = 0.0f;
        for (int j = 0; j < DMODEL; ++j) s += tile[r][j];
        float m  = s * (1.0f / DMODEL);
        float v2 = 0.0f;
        for (int j = 0; j < DMODEL; ++j) { float dd = tile[r][j] - m; v2 += dd * dd; }
        mu[r] = m;
        rs[r] = rsqrtf(v2 * (1.0f / DMODEL) + 1e-5f);
    }
    __syncthreads();

    int dim  = threadIdx.x & (DMODEL - 1);
    int half = threadIdx.x >> 7;
    float p = 0.0f;
#pragma unroll
    for (int r = half * 8; r < half * 8 + 8; ++r) p += (tile[r][dim] - mu[r]) * rs[r];
    p = p * lng[dim] + 8.0f * lnb[dim];
    part[threadIdx.x] = p;
    __syncthreads();

    if (threadIdx.x < DMODEL) {
        float tot = (part[threadIdx.x] + part[threadIdx.x + DMODEL]) * (1.0f / (float)L_SEQ);
        int b = mtile >> 8;
        atomicAdd(&out[b * DMODEL + dim], tot);
    }
}

// ---------------- host launcher ----------------
extern "C" void kernel_launch(void* const* d_in, const int* in_sizes, int n_in,
                              void* d_out, int out_size, void* d_ws, size_t ws_size,
                              hipStream_t stream) {
    const float* x        = (const float*)d_in[0];
    const float* conv_w   = (const float*)d_in[1];
    const float* conv_b   = (const float*)d_in[2];
    const float* bn_g     = (const float*)d_in[3];
    const float* bn_b     = (const float*)d_in[4];
    const float* bn_mean  = (const float*)d_in[5];
    const float* bn_var   = (const float*)d_in[6];
    const float* in_projw = (const float*)d_in[7];
    const float* dwc_w    = (const float*)d_in[8];
    const float* dwc_b    = (const float*)d_in[9];
    const float* x_projw  = (const float*)d_in[10];
    const float* dt_projw = (const float*)d_in[11];
    const float* dt_projb = (const float*)d_in[12];
    const float* A_log    = (const float*)d_in[13];
    const float* Dp       = (const float*)d_in[14];
    const float* out_projw= (const float*)d_in[15];
    const float* ln_g     = (const float*)d_in[16];
    const float* ln_b     = (const float*)d_in[17];
    float* out = (float*)d_out;

    uint8_t* ws = (uint8_t*)d_ws;
    size_t off = 0;
    auto carve = [&](size_t bytes) { void* p = ws + off; off += (bytes + 255) & ~(size_t)255; return p; };
    unsigned short* w_in  = (unsigned short*)carve((size_t)512 * 128 * 2);
    unsigned short* w_xp  = (unsigned short*)carve((size_t)40 * 256 * 2);
    unsigned short* w_out = (unsigned short*)carve((size_t)128 * 256 * 2);
    unsigned short* hA    = (unsigned short*)carve((size_t)NTOK * DMODEL * 2);
    unsigned short* U     = (unsigned short*)carve((size_t)NTOK * DINNER * 2);
    unsigned short* Z     = (unsigned short*)carve((size_t)NTOK * DINNER * 2);
    unsigned short* UC    = (unsigned short*)carve((size_t)NTOK * DINNER * 2);
    float*          XD    = (float*)carve((size_t)NTOK * XDBL_N * 4);
    unsigned short* Yg    = (unsigned short*)carve((size_t)NTOK * DINNER * 2);
    (void)ws_size; (void)in_sizes; (void)n_in; (void)out_size;

    k_wconv<<<256, 256, 0, stream>>>(in_projw, x_projw, out_projw, w_in, w_xp, w_out);

    {
        size_t tot = (size_t)NTOK * DMODEL;
        k_front<<<(unsigned)((tot + 255) / 256), 256, 0, stream>>>(
            x, conv_w, conv_b, bn_g, bn_b, bn_mean, bn_var, hA);
    }

    k_inproj<<<dim3(NTOK / 32, 2), 256, 0, stream>>>(hA, w_in, U, Z);

    {
        size_t tot = (size_t)NTOK * DINNER;
        k_dwconv<<<(unsigned)((tot + 255) / 256), 256, 0, stream>>>(U, dwc_w, dwc_b, UC);
    }

    k_xproj<<<NTOK / 16, 96, 0, stream>>>(UC, w_xp, XD);

    k_scan<<<B_SZ, DINNER, 0, stream>>>(XD, UC, Z, dt_projw, dt_projb, A_log, Dp, Yg);

    k_zero<<<8, 256, 0, stream>>>(out);

    k_outln<<<NTOK / 16, 256, 0, stream>>>(Yg, w_out, ln_g, ln_b, out);
}
